// TernaryLinear_89644557402776
// MI455X (gfx1250) — matrix operand order repaired and checked by reading, not compiled, not run on hardware
//
#include <hip/hip_runtime.h>

typedef _Float16 v16h __attribute__((ext_vector_type(16)));
typedef _Float16 v8h  __attribute__((ext_vector_type(8)));
typedef _Float16 v4h  __attribute__((ext_vector_type(4)));
typedef float    v8f  __attribute__((ext_vector_type(8)));
typedef float    v4f  __attribute__((ext_vector_type(4)));
typedef int      v4i  __attribute__((ext_vector_type(4)));

typedef __attribute__((address_space(1))) v4i* gbl_v4i_p;   // global
typedef __attribute__((address_space(3))) v4i* lds_v4i_p;   // LDS

constexpr int kTokens = 4096;
constexpr int kIn     = 4096;
constexpr int kOut    = 4096;
constexpr int kMaskW  = kIn / 8;

constexpr int BM = 128;
constexpr int BN = 128;

// ---------------------------------------------------------------------------
// CDNA5 async global->LDS copy (ASYNCcnt) with portable fallback
// ---------------------------------------------------------------------------
#if defined(__has_builtin)
#if __has_builtin(__builtin_amdgcn_global_load_async_to_lds_b128)
#define HAS_ASYNC_LDS 1
#endif
#if __has_builtin(__builtin_amdgcn_s_wait_asynccnt)
#define HAS_WAIT_ASYNC 1
#endif
#endif

__device__ __forceinline__ void async_cp16(const _Float16* gsrc, _Float16* ldst) {
#if defined(HAS_ASYNC_LDS)
    __builtin_amdgcn_global_load_async_to_lds_b128(
        (gbl_v4i_p)const_cast<_Float16*>(gsrc),
        (lds_v4i_p)ldst,
        0, 0);
#else
    *(v8h*)ldst = *(const v8h*)gsrc;
#endif
}

__device__ __forceinline__ void wait_async() {
#if defined(HAS_ASYNC_LDS)
#if defined(HAS_WAIT_ASYNC)
    __builtin_amdgcn_s_wait_asynccnt(0);
#else
    asm volatile("s_wait_asynccnt 0x0" ::: "memory");
#endif
#endif
}

// ============================================================================
// Pass 1a: x (f32) -> f16, once (memory bound)
// ============================================================================
__global__ __launch_bounds__(256)
void convert_x_f16(const float* __restrict__ x, _Float16* __restrict__ gx)
{
    const size_t u = (size_t)blockIdx.x * 256 + threadIdx.x;
    const float* src = x + u * 8;
    v4f a = *(const v4f*)(src);
    v4f b = *(const v4f*)(src + 4);
    v8h h;
    h[0] = (_Float16)a[0]; h[1] = (_Float16)a[1];
    h[2] = (_Float16)a[2]; h[3] = (_Float16)a[3];
    h[4] = (_Float16)b[0]; h[5] = (_Float16)b[1];
    h[6] = (_Float16)b[2]; h[7] = (_Float16)b[3];
    *(v8h*)(gx + u * 8) = h;
}

// ============================================================================
// Pass 1b: ternary masks -> f16 weights {-1,0,+1}, once (MSB-first bit order)
// ============================================================================
__global__ __launch_bounds__(256)
void decode_w_f16(const int* __restrict__ pos, const int* __restrict__ neg,
                  _Float16* __restrict__ gw)
{
    const size_t u = (size_t)blockIdx.x * 256 + threadIdx.x;
    const int p = pos[u];
    const int n = neg[u];
    v8h h;
#pragma unroll
    for (int t = 0; t < 8; ++t)
        h[t] = (_Float16)(((p >> (7 - t)) & 1) - ((n >> (7 - t)) & 1));
    *(v8h*)(gw + u * 8) = h;
}

// ============================================================================
// Pass 2: f16 WMMA GEMM, double-buffered async-to-LDS staging
// ============================================================================
constexpr int BK2     = 64;
constexpr int STRIDE2 = 72;   // halves; 144B rows -> conflict-free fragment reads
constexpr int TILEH   = BM * STRIDE2;

__global__ __launch_bounds__(256)
void wmma_gemm_f16(const _Float16* __restrict__ gx,
                   const _Float16* __restrict__ gw,
                   const float* __restrict__ scale_p,
                   const float* __restrict__ bias,
                   float* __restrict__ out)
{
    __shared__ __align__(16) _Float16 sA[2][TILEH];
    __shared__ __align__(16) _Float16 sB[2][TILEH];

    const int tid  = threadIdx.x;
    const int lane = tid & 31;
    const int wid  = tid >> 5;
    const int mw   = wid & 3;        // 4 waves along M (32 rows each)
    const int nw   = wid >> 2;       // 2 waves along N (64 cols each)
    const int kh   = lane >> 4;
    const int l16  = lane & 15;

    const int tBase = blockIdx.x * BM;
    const int oBase = blockIdx.y * BN;

    v8f acc[2][4];
#pragma unroll
    for (int i = 0; i < 2; ++i)
#pragma unroll
        for (int j = 0; j < 4; ++j)
            acc[i][j] = (v8f){0.f, 0.f, 0.f, 0.f, 0.f, 0.f, 0.f, 0.f};

    // staging slice: 1024 chunks of 8 halves (16B), 4 per thread
    const int row0 = tid >> 3;
    const int col0 = (tid & 7) * 8;
    const _Float16* gA = gx + (size_t)(tBase + row0) * kIn + col0;
    const _Float16* gB = gw + (size_t)(oBase + row0) * kIn + col0;

    auto issue_tile = [&](int buf, int k0) {
#pragma unroll
        for (int j = 0; j < 4; ++j) {
            const int row = row0 + 32 * j;
            async_cp16(gA + (size_t)(32 * j) * kIn + k0, &sA[buf][row * STRIDE2 + col0]);
            async_cp16(gB + (size_t)(32 * j) * kIn + k0, &sB[buf][row * STRIDE2 + col0]);
        }
    };

    // prologue: stage first tile
    issue_tile(0, 0);
    wait_async();
    __syncthreads();

    int buf = 0;
    for (int k0 = 0; k0 < kIn; k0 += BK2) {
        // overlap: start copying next tile into the other buffer
        if (k0 + BK2 < kIn)
            issue_tile(buf ^ 1, k0 + BK2);

        // compute on current buffer: two WMMA k-steps
#pragma unroll
        for (int kk = 0; kk < BK2; kk += 32) {
            v16h afrag[2];
#pragma unroll
            for (int i = 0; i < 2; ++i) {
                const int m = mw * 32 + i * 16 + l16;
                // A element e -> k = kk + 16*(e>>3) + 8*kh + (e&7)
                v8h lo = *(const v8h*)&sA[buf][m * STRIDE2 + kk + kh * 8];
                v8h hi = *(const v8h*)&sA[buf][m * STRIDE2 + kk + kh * 8 + 16];
                afrag[i] = __builtin_shufflevector(lo, hi, 0, 1, 2, 3, 4, 5, 6, 7,
                                                   8, 9, 10, 11, 12, 13, 14, 15);
            }
            v16h bfrag[4];
#pragma unroll
            for (int j = 0; j < 4; ++j) {
                const int o = nw * 64 + j * 16 + l16;
                // B element e -> k = kk + 16*(e>>3) + 8*kh + (e&7)
                v8h lo = *(const v8h*)&sB[buf][o * STRIDE2 + kk + kh * 8];
                v8h hi = *(const v8h*)&sB[buf][o * STRIDE2 + kk + kh * 8 + 16];
                bfrag[j] = __builtin_shufflevector(lo, hi, 0, 1, 2, 3, 4, 5, 6, 7,
                                                   8, 9, 10, 11, 12, 13, 14, 15);
            }
#pragma unroll
            for (int i = 0; i < 2; ++i)
#pragma unroll
                for (int j = 0; j < 4; ++j)
                    acc[i][j] = __builtin_amdgcn_wmma_f32_16x16x32_f16(
                        false, afrag[i], false, bfrag[j],
                        (short)0, acc[i][j], false, false);
        }

        // my async copies into buf^1 have landed; barrier makes them visible
        // to all waves and confirms everyone is done reading `buf`
        wait_async();
        __syncthreads();
        buf ^= 1;
    }

    const float s = *scale_p;
#pragma unroll
    for (int i = 0; i < 2; ++i) {
        const int mrow0 = tBase + mw * 32 + i * 16 + kh * 8;
#pragma unroll
        for (int j = 0; j < 4; ++j) {
            const int ocol = oBase + nw * 64 + j * 16 + l16;
            const float bv = bias[ocol];
#pragma unroll
            for (int r = 0; r < 8; ++r)
                out[(size_t)(mrow0 + r) * kOut + ocol] = s * acc[i][j][r] + bv;
        }
    }
}

// ============================================================================
// Fallback: fused decode-in-loop kernel (only if ws too small)
// ============================================================================
constexpr int STRIDEf = 40;

__global__ __launch_bounds__(256)
void ternary_wmma_fused(const float* __restrict__ x,
                        const int*   __restrict__ pos,
                        const int*   __restrict__ neg,
                        const float* __restrict__ scale_p,
                        const float* __restrict__ bias,
                        float*       __restrict__ out)
{
    __shared__ __align__(16) _Float16 sA[BM * STRIDEf];
    __shared__ __align__(16) _Float16 sB[BN * STRIDEf];

    const int tid  = threadIdx.x;
    const int lane = tid & 31;
    const int wid  = tid >> 5;
    const int mw   = wid & 3;
    const int nw   = wid >> 2;
    const int kh   = lane >> 4;
    const int l16  = lane & 15;

    const int tBase = blockIdx.x * BM;
    const int oBase = blockIdx.y * BN;

    v8f acc[2][4];
#pragma unroll
    for (int i = 0; i < 2; ++i)
#pragma unroll
        for (int j = 0; j < 4; ++j)
            acc[i][j] = (v8f){0.f, 0.f, 0.f, 0.f, 0.f, 0.f, 0.f, 0.f};

    const int oDec = tid >> 1;
    const int sub  = tid & 1;

    for (int k0 = 0; k0 < kIn; k0 += 32) {
#pragma unroll
        for (int j = 0; j < 4; ++j) {
            const int chunk = tid + 256 * j;
            const int row   = chunk >> 3;
            const int col   = (chunk & 7) * 4;
            v4f xv = *(const v4f*)(x + (size_t)(tBase + row) * kIn + k0 + col);
            v4h hv;
            hv[0] = (_Float16)xv[0]; hv[1] = (_Float16)xv[1];
            hv[2] = (_Float16)xv[2]; hv[3] = (_Float16)xv[3];
            *(v4h*)&sA[row * STRIDEf + col] = hv;
        }
        {
            const int kb = (k0 >> 3) + sub * 2;
            const int* pp = pos + (size_t)(oBase + oDec) * kMaskW + kb;
            const int* np = neg + (size_t)(oBase + oDec) * kMaskW + kb;
#pragma unroll
            for (int b = 0; b < 2; ++b) {
                const int p = pp[b];
                const int n = np[b];
                _Float16* dst = &sB[oDec * STRIDEf + sub * 16 + b * 8];
#pragma unroll
                for (int t = 0; t < 8; ++t)
                    dst[t] = (_Float16)(((p >> (7 - t)) & 1) - ((n >> (7 - t)) & 1));
            }
        }
        __syncthreads();

        v16h afrag[2];
#pragma unroll
        for (int i = 0; i < 2; ++i) {
            const int m = mw * 32 + i * 16 + l16;
            v8h lo = *(const v8h*)&sA[m * STRIDEf + kh * 8];
            v8h hi = *(const v8h*)&sA[m * STRIDEf + kh * 8 + 16];
            afrag[i] = __builtin_shufflevector(lo, hi, 0, 1, 2, 3, 4, 5, 6, 7,
                                               8, 9, 10, 11, 12, 13, 14, 15);
        }
        v16h bfrag[4];
#pragma unroll
        for (int j = 0; j < 4; ++j) {
            const int o = nw * 64 + j * 16 + l16;
            v8h lo = *(const v8h*)&sB[o * STRIDEf + kh * 8];
            v8h hi = *(const v8h*)&sB[o * STRIDEf + kh * 8 + 16];
            bfrag[j] = __builtin_shufflevector(lo, hi, 0, 1, 2, 3, 4, 5, 6, 7,
                                               8, 9, 10, 11, 12, 13, 14, 15);
        }
#pragma unroll
        for (int i = 0; i < 2; ++i)
#pragma unroll
            for (int j = 0; j < 4; ++j)
                acc[i][j] = __builtin_amdgcn_wmma_f32_16x16x32_f16(
                    false, afrag[i], false, bfrag[j],
                    (short)0, acc[i][j], false, false);
        __syncthreads();
    }

    const float s = *scale_p;
#pragma unroll
    for (int i = 0; i < 2; ++i) {
        const int mrow0 = tBase + mw * 32 + i * 16 + kh * 8;
#pragma unroll
        for (int j = 0; j < 4; ++j) {
            const int ocol = oBase + nw * 64 + j * 16 + l16;
            const float bv = bias[ocol];
#pragma unroll
            for (int r = 0; r < 8; ++r)
                out[(size_t)(mrow0 + r) * kOut + ocol] = s * acc[i][j][r] + bv;
        }
    }
}

// ============================================================================
extern "C" void kernel_launch(void* const* d_in, const int* in_sizes, int n_in,
                              void* d_out, int out_size, void* d_ws, size_t ws_size,
                              hipStream_t stream) {
    const float* x     = (const float*)d_in[0];
    const int*   pos   = (const int*)d_in[1];
    const int*   neg   = (const int*)d_in[2];
    const float* scale = (const float*)d_in[3];
    const float* bias  = (const float*)d_in[4];
    float*       out   = (float*)d_out;

    dim3 grid(kTokens / BM, kOut / BN);
    dim3 block(256);

    const size_t xBytes = (size_t)kTokens * kIn * sizeof(_Float16);
    const size_t wBytes = (size_t)kOut * kIn * sizeof(_Float16);

    if (ws_size >= xBytes + wBytes) {
        _Float16* gx = (_Float16*)d_ws;
        _Float16* gw = gx + (size_t)kTokens * kIn;
        convert_x_f16<<<(kTokens * (size_t)kIn / 8) / 256, 256, 0, stream>>>(x, gx);
        decode_w_f16<<<((size_t)kOut * kMaskW) / 256, 256, 0, stream>>>(pos, neg, gw);
        wmma_gemm_f16<<<grid, block, 0, stream>>>(gx, gw, scale, bias, out);
    } else {
        ternary_wmma_fused<<<grid, block, 0, stream>>>(x, pos, neg, scale, bias, out);
    }
}
